// Multi_Head_Attention_22222160789771
// MI455X (gfx1250) — matrix operand order repaired and checked by reading, not compiled, not run on hardware
//
#include <hip/hip_runtime.h>
#include <hip/hip_bf16.h>
#include <stdint.h>

// Problem constants (reference: B=4, S=2048, E=1024, H=16, DH=64)
#define S_LEN 2048
#define EMB   1024
#define NH    16
#define DHEAD 64
#define NB    4

typedef __attribute__((ext_vector_type(16))) __bf16 v16bf;
typedef __attribute__((ext_vector_type(8)))  float  v8f;
typedef __attribute__((__vector_size__(4 * sizeof(int)))) int ai4;

union Frag16 { uint4 q[2]; v16bf v; };

// ---- CDNA5 async global->LDS copy (ASYNCcnt path), with safe fallback ----
// Probe result (round 2 diagnostic): builtin exists; param0 is
// 'int4 __device__*' (addrspace(1)), so cast to the exact vector pointee.
#if defined(__HIP_DEVICE_COMPILE__) && \
    __has_builtin(__builtin_amdgcn_global_load_async_to_lds_b128) && \
    __has_builtin(__builtin_amdgcn_s_wait_asynccnt)
#define ASYNC_CP16(dst_lds, src_glb)                                      \
    __builtin_amdgcn_global_load_async_to_lds_b128(                       \
        (__attribute__((address_space(1))) ai4*)(src_glb),                \
        (__attribute__((address_space(3))) ai4*)(dst_lds), 0, 0)
#define ASYNC_FENCE() __builtin_amdgcn_s_wait_asynccnt(0)
#else
#define ASYNC_CP16(dst_lds, src_glb) \
    (*(uint4*)(dst_lds) = *(const uint4*)(src_glb))
#define ASYNC_FENCE() ((void)0)
#endif

__device__ __forceinline__ uint32_t cvt2_bf16(float a, float b) {
    // round-to-nearest-even f32 -> bf16, packed pair
    uint32_t ua = __float_as_uint(a);
    uint32_t ub = __float_as_uint(b);
    ua += 0x7FFFu + ((ua >> 16) & 1u);
    ub += 0x7FFFu + ((ub >> 16) & 1u);
    return (ua >> 16) | (ub & 0xFFFF0000u);
}

__device__ __forceinline__ uint16_t bf16_bits(float a) {
    uint32_t u = __float_as_uint(a);
    u += 0x7FFFu + ((u >> 16) & 1u);
    return (uint16_t)(u >> 16);
}

__device__ __forceinline__ v8f zero8() {
    v8f z = {0.f, 0.f, 0.f, 0.f, 0.f, 0.f, 0.f, 0.f};
    return z;
}

// A-matrix fragment (16x32 bf16): lane L holds row m=L&15; lanes 0-15 carry
// K=[0..7]+[16..23], lanes 16-31 carry K=[8..15]+[24..31] (ISA 7.12.2).
__device__ __forceinline__ void load_a_frag(Frag16& f, const __bf16* lds, int ld,
                                            int mbase, int kbase, int lane) {
    const __bf16* p = lds + (mbase + (lane & 15)) * ld + kbase + ((lane >> 4) << 3);
    f.q[0] = *(const uint4*)p;
    f.q[1] = *(const uint4*)(p + 16);
}

// B-matrix fragment (32x16 bf16): lane L holds column n=L&15; lanes 0-15 carry
// K=[0..7]+[16..23], lanes 16-31 K=[8..15]+[24..31] (same order as A).
__device__ __forceinline__ void load_b_frag(Frag16& f, const __bf16* lds, int ld,
                                            int nbase, int kbase, int lane) {
    const __bf16* p = lds + (nbase + (lane & 15)) * ld + kbase + ((lane >> 4) << 3);
    f.q[0] = *(const uint4*)p;
    f.q[1] = *(const uint4*)(p + 16);
}

__device__ __forceinline__ v8f wmma_bf16(const Frag16& a, const Frag16& b, v8f c) {
    return __builtin_amdgcn_wmma_f32_16x16x32_bf16(false, a.v, false, b.v,
                                                   (short)0, c, false, false);
}

// ---------------------------------------------------------------------------
// NT GEMM: D[m,n] = sum_k A[m,k] * W[n,k] + bias[n]
// block tile 128x128, BK=64, 8 waves (4x2), wave tile 32x64 (2x4 WMMA tiles).
// mode 0: Q -> bf16 [B,H,S,DH]; mode 1: K -> bf16 [B,H,S,DH];
// mode 2: V -> bf16 [B,H,DH,S] (transposed); mode 3: f32 out [M,E].
// ---------------------------------------------------------------------------
template <bool ABF>
__global__ __launch_bounds__(256) void gemm_nt_kernel(
    const void* __restrict__ Asrc, const float* __restrict__ W,
    const float* __restrict__ bias, void* __restrict__ Dst, int mode) {
    __shared__ __align__(16) __bf16 lds_a[128 * 72];
    __shared__ __align__(16) __bf16 lds_b[128 * 72];

    const int tid  = threadIdx.x;
    const int lane = tid & 31;
    const int wave = tid >> 5;
    const int wm   = (wave & 3) * 32;
    const int wn   = (wave >> 2) * 64;
    const int m0   = blockIdx.y * 128;
    const int n0   = blockIdx.x * 128;

    v8f acc[2][4];
#pragma unroll
    for (int i = 0; i < 2; ++i)
#pragma unroll
        for (int j = 0; j < 4; ++j) acc[i][j] = zero8();

    for (int k0 = 0; k0 < EMB; k0 += 64) {
        if (k0) __syncthreads();
        // ---- stage A tile (128 x 64) ----
        if (ABF) {
            const uint16_t* A = (const uint16_t*)Asrc;
#pragma unroll
            for (int it = 0; it < 4; ++it) {
                int id  = it * 256 + tid;
                int row = id >> 3;
                int c8  = (id & 7) << 3;
                ASYNC_CP16(lds_a + row * 72 + c8,
                           A + (size_t)(m0 + row) * EMB + k0 + c8);
            }
        } else {
            const float* A = (const float*)Asrc;
#pragma unroll
            for (int it = 0; it < 8; ++it) {
                int id  = it * 256 + tid;
                int row = id >> 4;
                int c4  = (id & 15) << 2;
                float4 fv = *(const float4*)(A + (size_t)(m0 + row) * EMB + k0 + c4);
                uint32_t* dst = (uint32_t*)(lds_a + row * 72 + c4);
                dst[0] = cvt2_bf16(fv.x, fv.y);
                dst[1] = cvt2_bf16(fv.z, fv.w);
            }
        }
        // ---- stage B tile (weight rows, 128 x 64, f32 -> bf16) ----
#pragma unroll
        for (int it = 0; it < 8; ++it) {
            int id  = it * 256 + tid;
            int row = id >> 4;
            int c4  = (id & 15) << 2;
            float4 fv = *(const float4*)(W + (size_t)(n0 + row) * EMB + k0 + c4);
            uint32_t* dst = (uint32_t*)(lds_b + row * 72 + c4);
            dst[0] = cvt2_bf16(fv.x, fv.y);
            dst[1] = cvt2_bf16(fv.z, fv.w);
        }
        if (ABF) ASYNC_FENCE();
        __syncthreads();
        // ---- compute: 2 k-steps of 32, 8 WMMAs each ----
#pragma unroll
        for (int ks = 0; ks < 2; ++ks) {
            Frag16 af[2];
            Frag16 bf_[4];
#pragma unroll
            for (int mt = 0; mt < 2; ++mt)
                load_a_frag(af[mt], lds_a, 72, wm + mt * 16, ks * 32, lane);
#pragma unroll
            for (int nt = 0; nt < 4; ++nt)
                load_b_frag(bf_[nt], lds_b, 72, wn + nt * 16, ks * 32, lane);
#pragma unroll
            for (int mt = 0; mt < 2; ++mt)
#pragma unroll
                for (int nt = 0; nt < 4; ++nt)
                    acc[mt][nt] = wmma_bf16(af[mt], bf_[nt], acc[mt][nt]);
        }
    }

    // ---- epilogue: C/D layout row = r + 8*(lane>=16), col = lane&15 ----
    const int rowsel = (lane >> 4) << 3;
    const int colsel = lane & 15;
#pragma unroll
    for (int mt = 0; mt < 2; ++mt) {
#pragma unroll
        for (int nt = 0; nt < 4; ++nt) {
#pragma unroll
            for (int r = 0; r < 8; ++r) {
                int m = m0 + wm + mt * 16 + r + rowsel;
                int n = n0 + wn + nt * 16 + colsel;
                float val = acc[mt][nt][r] + bias[n];
                if (mode == 3) {
                    ((float*)Dst)[(size_t)m * EMB + n] = val;
                } else {
                    int b = m >> 11, s = m & (S_LEN - 1);
                    int h = n >> 6, d = n & (DHEAD - 1);
                    size_t idx;
                    if (mode == 2)
                        idx = (((size_t)(b * NH + h)) * DHEAD + d) * S_LEN + s;
                    else
                        idx = (((size_t)(b * NH + h)) * S_LEN + s) * DHEAD + d;
                    ((uint16_t*)Dst)[idx] = bf16_bits(val);
                }
            }
        }
    }
}

// ---------------------------------------------------------------------------
// scores[b,h,i,j] = (Q[b,h,i,:] . K[b,h,j,:]) / 8.
// Fully masked 128x128 tiles are skipped (softmax zero-fills them); only
// diagonal tiles pay the per-element mask compare.
// ---------------------------------------------------------------------------
__global__ __launch_bounds__(256) void score_kernel(
    const uint16_t* __restrict__ Qb, const uint16_t* __restrict__ Kb,
    float* __restrict__ attn, const int* __restrict__ causalp) {
    __shared__ __align__(16) __bf16 lds_q[128 * 72];
    __shared__ __align__(16) __bf16 lds_k[128 * 72];

    const int causal = *causalp;
    const int bx = blockIdx.x, by = blockIdx.y, bh = blockIdx.z;
    if (causal && bx > by) return;  // tile entirely above the diagonal

    const int tid  = threadIdx.x;
    const int lane = tid & 31;
    const int wave = tid >> 5;
    const int wm   = (wave & 3) * 32;
    const int wn   = (wave >> 2) * 64;
    const int m0   = by * 128;
    const int n0   = bx * 128;

    const uint16_t* Q = Qb + (size_t)bh * S_LEN * DHEAD;
    const uint16_t* K = Kb + (size_t)bh * S_LEN * DHEAD;

#pragma unroll
    for (int it = 0; it < 4; ++it) {
        int id  = it * 256 + tid;
        int row = id >> 3;
        int c8  = (id & 7) << 3;
        ASYNC_CP16(lds_q + row * 72 + c8, Q + (size_t)(m0 + row) * DHEAD + c8);
        ASYNC_CP16(lds_k + row * 72 + c8, K + (size_t)(n0 + row) * DHEAD + c8);
    }
    ASYNC_FENCE();
    __syncthreads();

    v8f acc[2][4];
#pragma unroll
    for (int i = 0; i < 2; ++i)
#pragma unroll
        for (int j = 0; j < 4; ++j) acc[i][j] = zero8();

#pragma unroll
    for (int ks = 0; ks < 2; ++ks) {
        Frag16 af[2];
        Frag16 bf_[4];
#pragma unroll
        for (int mt = 0; mt < 2; ++mt)
            load_a_frag(af[mt], lds_q, 72, wm + mt * 16, ks * 32, lane);
#pragma unroll
        for (int nt = 0; nt < 4; ++nt)
            load_b_frag(bf_[nt], lds_k, 72, wn + nt * 16, ks * 32, lane);
#pragma unroll
        for (int mt = 0; mt < 2; ++mt)
#pragma unroll
            for (int nt = 0; nt < 4; ++nt)
                acc[mt][nt] = wmma_bf16(af[mt], bf_[nt], acc[mt][nt]);
    }

    float* base = attn + (size_t)bh * S_LEN * S_LEN;
    const int rowsel = (lane >> 4) << 3;
    const int colsel = lane & 15;
    const bool diag  = (causal != 0) && (bx == by);
    if (!diag) {
        // fully-unmasked tile: plain scaled stores, no per-element compares
#pragma unroll
        for (int mt = 0; mt < 2; ++mt)
#pragma unroll
            for (int nt = 0; nt < 4; ++nt)
#pragma unroll
                for (int r = 0; r < 8; ++r) {
                    int i = m0 + wm + mt * 16 + r + rowsel;
                    int j = n0 + wn + nt * 16 + colsel;
                    base[(size_t)i * S_LEN + j] = acc[mt][nt][r] * 0.125f;
                }
    } else {
#pragma unroll
        for (int mt = 0; mt < 2; ++mt)
#pragma unroll
            for (int nt = 0; nt < 4; ++nt)
#pragma unroll
                for (int r = 0; r < 8; ++r) {
                    int i = m0 + wm + mt * 16 + r + rowsel;
                    int j = n0 + wn + nt * 16 + colsel;
                    float val = acc[mt][nt][r] * 0.125f;
                    if (j > i) val = -1e30f;
                    base[(size_t)i * S_LEN + j] = val;
                }
    }
}

// ---------------------------------------------------------------------------
// Row softmax over last dim (length 2048), float4-vectorized; masks by index
// so it never reads the skipped upper-triangle region; writes exact zeros.
// ---------------------------------------------------------------------------
__global__ __launch_bounds__(256) void softmax_kernel(float* __restrict__ attn,
                                                      const int* __restrict__ causalp) {
    const int row = blockIdx.x;  // b*H*S rows
    const int i   = row & (S_LEN - 1);
    const int L   = (*causalp) ? (i + 1) : S_LEN;
    float* p = attn + (size_t)row * S_LEN;
    const int tid = threadIdx.x;
    __shared__ float red[256];

    float x[8];
    float mx = -3.0e38f;
#pragma unroll
    for (int t = 0; t < 2; ++t) {
        int j = (t * 256 + tid) << 2;
        float4 v;
        if (j < L) {
            // partial groups stay inside the written diagonal tile => safe read
            v = *(const float4*)(p + j);
        } else {
            v = make_float4(-3.0e38f, -3.0e38f, -3.0e38f, -3.0e38f);
        }
        float c0 = (j + 0 < L) ? v.x : -3.0e38f;
        float c1 = (j + 1 < L) ? v.y : -3.0e38f;
        float c2 = (j + 2 < L) ? v.z : -3.0e38f;
        float c3 = (j + 3 < L) ? v.w : -3.0e38f;
        x[t * 4 + 0] = c0; x[t * 4 + 1] = c1; x[t * 4 + 2] = c2; x[t * 4 + 3] = c3;
        mx = fmaxf(mx, fmaxf(fmaxf(c0, c1), fmaxf(c2, c3)));
    }
    red[tid] = mx;
    __syncthreads();
    for (int s = 128; s > 0; s >>= 1) {
        if (tid < s) red[tid] = fmaxf(red[tid], red[tid + s]);
        __syncthreads();
    }
    mx = red[0];
    __syncthreads();

    float sum = 0.f;
#pragma unroll
    for (int t = 0; t < 8; ++t) {
        int j   = ((t >> 2) * 256 + tid) * 4 + (t & 3);
        float e = (j < L) ? __expf(x[t] - mx) : 0.f;
        x[t] = e;
        sum += e;
    }
    red[tid] = sum;
    __syncthreads();
    for (int s = 128; s > 0; s >>= 1) {
        if (tid < s) red[tid] += red[tid + s];
        __syncthreads();
    }
    float inv = 1.0f / red[0];
#pragma unroll
    for (int t = 0; t < 2; ++t) {
        int j = (t * 256 + tid) << 2;
        float4 o = make_float4(x[t * 4 + 0] * inv, x[t * 4 + 1] * inv,
                               x[t * 4 + 2] * inv, x[t * 4 + 3] * inv);
        *(float4*)(p + j) = o;
    }
}

// ---------------------------------------------------------------------------
// O[b,h,i,d] = sum_j P[b,h,i,j] * V[b,h,j,d]; V stored transposed [B,H,DH,S]
// so the B fragment is contiguous. P converted f32->bf16 on the fly.
// Causal truncates the K-loop at the diagonal block.
// ---------------------------------------------------------------------------
__global__ __launch_bounds__(256) void pv_kernel(
    const float* __restrict__ attn, const uint16_t* __restrict__ Vt,
    uint16_t* __restrict__ Oh, const int* __restrict__ causalp) {
    __shared__ __align__(16) __bf16 lds_p[128 * 72];
    __shared__ __align__(16) __bf16 lds_v[64 * 72];

    const int tid  = threadIdx.x;
    const int lane = tid & 31;
    const int wave = tid >> 5;
    const int bh   = blockIdx.z;
    const int b    = bh >> 4;
    const int h    = bh & 15;
    const int m0   = blockIdx.y * 128;

    const float* P    = attn + (size_t)bh * S_LEN * S_LEN;
    const uint16_t* V = Vt + (size_t)bh * DHEAD * S_LEN;
    const int causal  = *causalp;
    const int kend    = causal ? (m0 + 128) : S_LEN;

    v8f acc[4];
#pragma unroll
    for (int j = 0; j < 4; ++j) acc[j] = zero8();

    for (int k0 = 0; k0 < kend; k0 += 64) {
        if (k0) __syncthreads();
        // stage Vt tile (64 rows(d) x 64 cols(j)) via async copy
#pragma unroll
        for (int it = 0; it < 2; ++it) {
            int id  = it * 256 + tid;
            int row = id >> 3;
            int c8  = (id & 7) << 3;
            ASYNC_CP16(lds_v + row * 72 + c8, V + (size_t)row * S_LEN + k0 + c8);
        }
        // stage P tile (128 x 64 f32 -> bf16, register conversion path)
#pragma unroll
        for (int it = 0; it < 8; ++it) {
            int id  = it * 256 + tid;
            int row = id >> 4;
            int c4  = (id & 15) << 2;
            float4 fv = *(const float4*)(P + (size_t)(m0 + row) * S_LEN + k0 + c4);
            uint32_t* dst = (uint32_t*)(lds_p + row * 72 + c4);
            dst[0] = cvt2_bf16(fv.x, fv.y);
            dst[1] = cvt2_bf16(fv.z, fv.w);
        }
        ASYNC_FENCE();
        __syncthreads();
#pragma unroll
        for (int ks = 0; ks < 2; ++ks) {
            Frag16 a;
            load_a_frag(a, lds_p, 72, wave * 16, ks * 32, lane);
#pragma unroll
            for (int nt = 0; nt < 4; ++nt) {
                Frag16 bfr;
                load_b_frag(bfr, lds_v, 72, nt * 16, ks * 32, lane);
                acc[nt] = wmma_bf16(a, bfr, acc[nt]);
            }
        }
    }

    const int rowsel = (lane >> 4) << 3;
    const int colsel = lane & 15;
    uint16_t* Ob = Oh + (size_t)(b * S_LEN + m0 + wave * 16 + rowsel) * EMB + h * DHEAD + colsel;
#pragma unroll
    for (int nt = 0; nt < 4; ++nt)
#pragma unroll
        for (int r = 0; r < 8; ++r)
            Ob[(size_t)r * EMB + nt * 16] = bf16_bits(acc[nt][r]);
}

// ---------------------------------------------------------------------------
extern "C" void kernel_launch(void* const* d_in, const int* in_sizes, int n_in,
                              void* d_out, int out_size, void* d_ws, size_t ws_size,
                              hipStream_t stream) {
    const float* q  = (const float*)d_in[0];
    const float* k  = (const float*)d_in[1];
    const float* v  = (const float*)d_in[2];
    const float* wq = (const float*)d_in[3];
    const float* bq = (const float*)d_in[4];
    const float* wk = (const float*)d_in[5];
    const float* bk = (const float*)d_in[6];
    const float* wv = (const float*)d_in[7];
    const float* bv = (const float*)d_in[8];
    const float* wo = (const float*)d_in[9];
    const float* bo = (const float*)d_in[10];
    const int* causal = (const int*)d_in[11];

    float* out  = (float*)d_out;
    float* attn = out + (size_t)NB * S_LEN * EMB;  // second tuple element

    const size_t MTOT = (size_t)NB * S_LEN * EMB;  // 8192*1024
    uint16_t* Qb = (uint16_t*)d_ws;
    uint16_t* Kb = Qb + MTOT;
    uint16_t* Vt = Kb + MTOT;
    uint16_t* Oh = Vt + MTOT;

    dim3 blk(256);
    dim3 gproj(EMB / 128, (NB * S_LEN) / 128);  // (8, 64)

    gemm_nt_kernel<false><<<gproj, blk, 0, stream>>>(q, wq, bq, Qb, 0);
    gemm_nt_kernel<false><<<gproj, blk, 0, stream>>>(k, wk, bk, Kb, 1);
    gemm_nt_kernel<false><<<gproj, blk, 0, stream>>>(v, wv, bv, Vt, 2);

    score_kernel<<<dim3(S_LEN / 128, S_LEN / 128, NB * NH), blk, 0, stream>>>(Qb, Kb, attn, causal);
    softmax_kernel<<<dim3(NB * NH * S_LEN), blk, 0, stream>>>(attn, causal);
    pv_kernel<<<dim3(1, S_LEN / 128, NB * NH), blk, 0, stream>>>(attn, Vt, Oh, causal);

    gemm_nt_kernel<true><<<gproj, blk, 0, stream>>>(Oh, wo, bo, out, 3);
}
